// GAT_29102698397788
// MI455X (gfx1250) — matrix operand order repaired and checked by reading, not compiled, not run on hardware
//
#include <hip/hip_runtime.h>
#include <hip/hip_bf16.h>

// ---------------------------------------------------------------------------
// GAT pipeline for MI455X (gfx1250, wave32, WMMA).
// GEMMs: v_wmma_f32_16x16x32_f16, f16 A/B, f32 accum, 2x2 tiles per wave,
// weights pre-swizzled into per-lane fragment order (2x b128 loads per frag).
// Edge softmax: 3-pass (atomicMax ordered-int, atomicAdd f32, scatter).
// Workspace requirement: ~154 MB.
// ---------------------------------------------------------------------------

typedef __attribute__((ext_vector_type(16))) _Float16 v16h;
typedef __attribute__((ext_vector_type(8)))  float    v8f;

#define NN     32768      // total nodes
#define GG     512        // graphs
#define EE     262144     // edges
#define HH     128        // hidden
#define HEADS1 4

// ------------------------- WMMA fragment helpers ---------------------------
// A fragment: 16x32 f16. lane<16: row=lane, K={0..7,16..23}; lane>=16: row=lane-16,
// K={8..15,24..31}.
__device__ inline v16h load_a_f32(const float* __restrict__ A, int lda) {
  int lane = threadIdx.x & 31;
  const float* p = A + (size_t)(lane & 15) * lda + ((lane >> 4) << 3);
  v16h a;
#pragma unroll
  for (int i = 0; i < 8; ++i) a[i] = (_Float16)p[i];
#pragma unroll
  for (int i = 0; i < 8; ++i) a[8 + i] = (_Float16)p[16 + i];
  return a;
}

__device__ inline v16h load_a_f16_gather(const _Float16* __restrict__ base,
                                         const int* __restrict__ ridx, int lda) {
  int lane = threadIdx.x & 31;
  const _Float16* p = base + (size_t)ridx[lane & 15] * lda + ((lane >> 4) << 3);
  v16h a;
#pragma unroll
  for (int i = 0; i < 8; ++i) a[i] = p[i];
#pragma unroll
  for (int i = 0; i < 8; ++i) a[8 + i] = p[16 + i];
  return a;
}

// B fragment from pre-swizzled storage: 16 contiguous halves per lane (2x b128).
// Swizzled layout: F[((kt*ntiles + nt)*32 + lane)*16 + i]
__device__ inline v16h load_b_swz(const _Float16* __restrict__ F) {
  int lane = threadIdx.x & 31;
  return *(const v16h*)(F + (size_t)lane * 16);
}

__device__ inline v8f wmma16(v16h a, v16h b, v8f c) {
  return __builtin_amdgcn_wmma_f32_16x16x32_f16(false, a, false, b,
                                                (short)0, c, false, false);
}

__device__ inline float wsum(float x) {
#pragma unroll
  for (int m = 16; m; m >>= 1) x += __shfl_xor(x, m, 32);
  return x;
}

__device__ inline int   fenc(float f) { int i = __float_as_int(f); return i >= 0 ? i : (i ^ 0x7fffffff); }
__device__ inline float fdec(int i)   { return __int_as_float(i >= 0 ? i : (i ^ 0x7fffffff)); }
__device__ inline float lrelu2(float x) { return x > 0.f ? x : 0.2f * x; }

// ------------------------------ kernels ------------------------------------
__global__ void k_cvt(const float* __restrict__ s, _Float16* __restrict__ d, int n) {
  int t = blockIdx.x * blockDim.x + threadIdx.x;
  if (t < n) d[t] = (_Float16)s[t];
}

// Convert row-major f32 weight [K,Ncols] into swizzled f16 fragment order:
// out[((kt*ntiles + nt)*32 + lane)*16 + i], lane<16 -> K=kt*32+{0..7,16..23}[i], col=nt*16+lane;
// lane>=16 -> K=kt*32+{8..15,24..31}[i].
__global__ void k_cvt_b(const float* __restrict__ W, _Float16* __restrict__ F,
                        int Ncols, int K) {
  int t = blockIdx.x * blockDim.x + threadIdx.x;
  if (t >= K * Ncols) return;
  int i    = t & 15;
  int lane = (t >> 4) & 31;
  int tile = t >> 9;                 // kt*ntiles + nt
  int ntiles = Ncols >> 4;
  int nt = tile % ntiles, kt = tile / ntiles;
  int col  = nt * 16 + (lane & 15);
  int krow = kt * 32 + ((i >> 3) << 4) + ((lane >> 4) << 3) + (i & 7);
  F[t] = (_Float16)W[(size_t)krow * Ncols + col];
}

// C[M,N] = A[M,K] @ Bswz[K,N] (+bias, act). One wave per 32x32 output (2x2 tiles).
__global__ void k_gemm(const float* __restrict__ A, const _Float16* __restrict__ Bswz,
                       const float* __restrict__ bias, float* __restrict__ C,
                       int Ncols, int K, int act) {
  int tm = blockIdx.x * 2, tn = blockIdx.y * 2;
  int ntiles = Ncols >> 4;
  const float* A0 = A + (size_t)tm * 16 * K;
  const float* A1 = A0 + (size_t)16 * K;
  v8f a00 = {}, a01 = {}, a10 = {}, a11 = {};
  for (int kt = 0; kt < (K >> 5); ++kt) {
    v16h fa0 = load_a_f32(A0 + kt * 32, K);
    v16h fa1 = load_a_f32(A1 + kt * 32, K);
    v16h fb0 = load_b_swz(Bswz + ((size_t)(kt * ntiles + tn) * 32) * 16);
    v16h fb1 = load_b_swz(Bswz + ((size_t)(kt * ntiles + tn + 1) * 32) * 16);
    a00 = wmma16(fa0, fb0, a00);
    a01 = wmma16(fa0, fb1, a01);
    a10 = wmma16(fa1, fb0, a10);
    a11 = wmma16(fa1, fb1, a11);
  }
  int lane = threadIdx.x & 31;
  int colo = lane & 15;
  int rsel = (lane >> 4) << 3;
  v8f accs[4] = {a00, a01, a10, a11};
#pragma unroll
  for (int q = 0; q < 4; ++q) {
    int tmi = tm + (q >> 1), tni = tn + (q & 1);
    int col = tni * 16 + colo;
    float bb = bias ? bias[col] : 0.f;
    int rb = tmi * 16 + rsel;
#pragma unroll
    for (int j = 0; j < 8; ++j) {
      float y = accs[q][j] + bb;
      if (act == 2) y = y > 0.f ? y : 0.01f * y;
      C[(size_t)(rb + j) * Ncols + col] = y;
    }
  }
}

// X[node,:] = emb[m_idx[node]] @ Wnode + bnode   (gathered-A WMMA)
__global__ void k_embed(const _Float16* __restrict__ embH, const int* __restrict__ midx,
                        const _Float16* __restrict__ WswH, const float* __restrict__ bias,
                        float* __restrict__ X) {
  int tm = blockIdx.x, tn = blockIdx.y;      // node tile, H tile (ntiles=8)
  const int* ridx = midx + tm * 16;
  v8f acc = {};
#pragma unroll
  for (int kt = 0; kt < 4; ++kt)
    acc = wmma16(load_a_f16_gather(embH + kt * 32, ridx, 128),
                 load_b_swz(WswH + ((size_t)(kt * 8 + tn) * 32) * 16), acc);
  int lane = threadIdx.x & 31;
  int col  = tn * 16 + (lane & 15);
  float bb = bias[col];
  int rb   = tm * 16 + ((lane >> 4) << 3);
#pragma unroll
  for (int j = 0; j < 8; ++j)
    X[(size_t)(rb + j) * 128 + col] = acc[j] + bb;
}

// te/ve encoder tiles into LDS, then start = [ve|te] @ Wstart, written to node g*64.
__global__ void k_start(const float* __restrict__ t, const float* __restrict__ v,
                        const _Float16* __restrict__ WtS, const float* __restrict__ bt,
                        const _Float16* __restrict__ WvS, const float* __restrict__ bv,
                        const _Float16* __restrict__ WstS, const float* __restrict__ bst,
                        float* __restrict__ X) {
  __shared__ float ld[16 * 256];
  int g0 = blockIdx.x * 16;
  int lane = threadIdx.x & 31;
  int col = lane & 15, rb = (lane >> 4) << 3;
  for (int nt = 0; nt < 8; ++nt) {
    v8f accv = {}, acct = {};
    for (int kt = 0; kt < 24; ++kt) {            // K=768
      v16h fav = load_a_f32(v + (size_t)g0 * 768 + kt * 32, 768);
      v16h fat = load_a_f32(t + (size_t)g0 * 768 + kt * 32, 768);
      accv = wmma16(fav, load_b_swz(WvS + ((size_t)(kt * 8 + nt) * 32) * 16), accv);
      acct = wmma16(fat, load_b_swz(WtS + ((size_t)(kt * 8 + nt) * 32) * 16), acct);
    }
    float bvv = bv[nt * 16 + col], btv = bt[nt * 16 + col];
#pragma unroll
    for (int j = 0; j < 8; ++j) {
      ld[(rb + j) * 256 + nt * 16 + col]       = accv[j] + bvv;   // ve -> cols [0,128)
      ld[(rb + j) * 256 + 128 + nt * 16 + col] = acct[j] + btv;   // te -> cols [128,256)
    }
  }
  __syncthreads();
  for (int nt = 0; nt < 8; ++nt) {
    v8f acc = {};
#pragma unroll
    for (int kt = 0; kt < 8; ++kt)               // K=256
      acc = wmma16(load_a_f32(ld + kt * 32, 256),
                   load_b_swz(WstS + ((size_t)(kt * 8 + nt) * 32) * 16), acc);
    float bsv = bst[nt * 16 + col];
#pragma unroll
    for (int j = 0; j < 8; ++j)
      X[(size_t)(g0 + rb + j) * 64 * 128 + nt * 16 + col] = acc[j] + bsv;  // node = g*64
  }
}

__global__ void k_ln(float* __restrict__ X, const float* __restrict__ g,
                     const float* __restrict__ b) {
  int n = blockIdx.x * 8 + (threadIdx.x >> 5);
  int lane = threadIdx.x & 31;
  float vv[4]; float s = 0.f;
#pragma unroll
  for (int i = 0; i < 4; ++i) { vv[i] = X[(size_t)n * 128 + lane + 32 * i]; s += vv[i]; }
  float mu = wsum(s) * (1.f / 128.f);
  float q = 0.f;
#pragma unroll
  for (int i = 0; i < 4; ++i) { float d = vv[i] - mu; q += d * d; }
  float rs = rsqrtf(wsum(q) * (1.f / 128.f) + 1e-6f);
#pragma unroll
  for (int i = 0; i < 4; ++i) {
    int c = lane + 32 * i;
    X[(size_t)n * 128 + c] = (vv[i] - mu) * rs * g[c] + b[c];
  }
}

template <int HEADS>
__global__ void k_dots(const float* __restrict__ XP, const float* __restrict__ aS,
                       const float* __restrict__ aD, float* __restrict__ als,
                       float* __restrict__ ald) {
  int n = blockIdx.x * 8 + (threadIdx.x >> 5);
  int lane = threadIdx.x & 31;
#pragma unroll
  for (int h = 0; h < HEADS; ++h) {
    float ps = 0.f, pd = 0.f;
#pragma unroll
    for (int i = 0; i < 4; ++i) {
      float xv = XP[(size_t)n * HEADS * 128 + h * 128 + lane + 32 * i];
      ps += xv * aS[h * 128 + lane + 32 * i];
      pd += xv * aD[h * 128 + lane + 32 * i];
    }
    ps = wsum(ps); pd = wsum(pd);
    if (lane == 0) { als[n * HEADS + h] = ps; ald[n * HEADS + h] = pd; }
  }
}

template <int HEADS>
__global__ void k_init_max(const float* __restrict__ als, const float* __restrict__ ald,
                           int* __restrict__ mx) {
  int t = blockIdx.x * blockDim.x + threadIdx.x;         // over N*HEADS
  mx[t] = fenc(lrelu2(als[t] + ald[t]));                 // self-loop score
}

template <int HEADS>
__global__ void k_edge_max(const int* __restrict__ src, const int* __restrict__ dst,
                           const float* __restrict__ als, const float* __restrict__ ald,
                           int* __restrict__ mx) {
  int e = blockIdx.x * blockDim.x + threadIdx.x;
  int s = src[e], d = dst[e];
#pragma unroll
  for (int h = 0; h < HEADS; ++h)
    atomicMax(&mx[d * HEADS + h], fenc(lrelu2(als[s * HEADS + h] + ald[d * HEADS + h])));
}

template <int HEADS>
__global__ void k_init_sum(const float* __restrict__ als, const float* __restrict__ ald,
                           const int* __restrict__ mx, float* __restrict__ sm) {
  int t = blockIdx.x * blockDim.x + threadIdx.x;
  sm[t] = expf(lrelu2(als[t] + ald[t]) - fdec(mx[t]));
}

template <int HEADS>
__global__ void k_edge_sum(const int* __restrict__ src, const int* __restrict__ dst,
                           const float* __restrict__ als, const float* __restrict__ ald,
                           const int* __restrict__ mx, float* __restrict__ sm) {
  int e = blockIdx.x * blockDim.x + threadIdx.x;
  int s = src[e], d = dst[e];
#pragma unroll
  for (int h = 0; h < HEADS; ++h) {
    float sc = lrelu2(als[s * HEADS + h] + ald[d * HEADS + h]);
    atomicAdd(&sm[d * HEADS + h], expf(sc - fdec(mx[d * HEADS + h])));
  }
}

template <int HEADS>
__global__ void k_init_acc(const float* __restrict__ XP, const float* __restrict__ als,
                           const float* __restrict__ ald, const int* __restrict__ mx,
                           const float* __restrict__ sm, float* __restrict__ ACC) {
  int n = blockIdx.x * 8 + (threadIdx.x >> 5);
  int lane = threadIdx.x & 31;
#pragma unroll
  for (int h = 0; h < HEADS; ++h) {
    float a = expf(lrelu2(als[n * HEADS + h] + ald[n * HEADS + h]) - fdec(mx[n * HEADS + h]))
              / sm[n * HEADS + h];
#pragma unroll
    for (int i = 0; i < 4; ++i) {
      size_t idx = (size_t)n * HEADS * 128 + h * 128 + lane + 32 * i;
      ACC[idx] = XP[idx] * a;
    }
  }
}

template <int HEADS>
__global__ void k_edge_scatter(const int* __restrict__ src, const int* __restrict__ dst,
                               const float* __restrict__ XP, const float* __restrict__ als,
                               const float* __restrict__ ald, const int* __restrict__ mx,
                               const float* __restrict__ sm, float* __restrict__ ACC) {
  int e = blockIdx.x * 8 + (threadIdx.x >> 5);
  int lane = threadIdx.x & 31;
  int s = src[e], d = dst[e];
  __builtin_prefetch(XP + (size_t)s * HEADS * 128 + lane * 4, 0, 0);  // global_prefetch_b8
#pragma unroll
  for (int h = 0; h < HEADS; ++h) {
    float a = expf(lrelu2(als[s * HEADS + h] + ald[d * HEADS + h]) - fdec(mx[d * HEADS + h]))
              / sm[d * HEADS + h];
#pragma unroll
    for (int i = 0; i < 4; ++i)
      atomicAdd(&ACC[(size_t)d * HEADS * 128 + h * 128 + lane + 32 * i],
                XP[(size_t)s * HEADS * 128 + h * 128 + lane + 32 * i] * a);
  }
}

__global__ void k_bias_elu(float* __restrict__ Y, const float* __restrict__ bias,
                           int width, int n) {
  int t = blockIdx.x * blockDim.x + threadIdx.x;
  if (t >= n) return;
  float y = Y[t] + bias[t & (width - 1)];
  Y[t] = y > 0.f ? y : expm1f(y);
}

__global__ void k_pool(const float* __restrict__ Xf, const int* __restrict__ midx,
                       float* __restrict__ P) {
  int g = blockIdx.x * 8 + (threadIdx.x >> 5);
  int lane = threadIdx.x & 31;
  float cnt = 0.f;
  for (int i = lane; i < 64; i += 32) cnt += (midx[g * 64 + i] >= 1) ? 1.f : 0.f;
  cnt = wsum(cnt);
  float acc[4] = {0.f, 0.f, 0.f, 0.f};
  for (int nn = 0; nn < 64; ++nn) {
    if (midx[g * 64 + nn] >= 1) {
#pragma unroll
      for (int i = 0; i < 4; ++i)
        acc[i] += Xf[(size_t)(g * 64 + nn) * 128 + lane + 32 * i];
    }
  }
  float inv = 1.f / cnt;
#pragma unroll
  for (int i = 0; i < 4; ++i) P[(size_t)g * 128 + lane + 32 * i] = acc[i] * inv;
}

__global__ void k_logits(const float* __restrict__ H1a, const float* __restrict__ Wo2,
                         const float* __restrict__ bo2, float* __restrict__ out) {
  int r = blockIdx.x * 8 + (threadIdx.x >> 5);
  int lane = threadIdx.x & 31;
  float p = 0.f;
#pragma unroll
  for (int i = 0; i < 4; ++i)
    p += H1a[(size_t)r * 128 + lane + 32 * i] * Wo2[lane + 32 * i];
  p = wsum(p);
  if (lane == 0) out[r] = p + bo2[0];
}

// ------------------------------ launcher -----------------------------------
extern "C" void kernel_launch(void* const* d_in, const int* in_sizes, int n_in,
                              void* d_out, int out_size, void* d_ws, size_t ws_size,
                              hipStream_t stream) {
  (void)in_sizes; (void)n_in; (void)out_size; (void)ws_size;

  const float* t      = (const float*)d_in[0];
  const float* v      = (const float*)d_in[1];
  const int*   m_idx  = (const int*)d_in[2];
  const int*   e_src  = (const int*)d_in[3];
  const int*   e_dst  = e_src + EE;
  const float* Wt     = (const float*)d_in[4];
  const float* bt     = (const float*)d_in[5];
  const float* Wv     = (const float*)d_in[6];
  const float* bv     = (const float*)d_in[7];
  const float* emb    = (const float*)d_in[8];
  const float* Wnode  = (const float*)d_in[9];
  const float* bnode  = (const float*)d_in[10];
  const float* Wstart = (const float*)d_in[11];
  const float* bstart = (const float*)d_in[12];
  const float* ln_g   = (const float*)d_in[13];
  const float* ln_b   = (const float*)d_in[14];
  const float* W1     = (const float*)d_in[15];
  const float* a_src1 = (const float*)d_in[16];
  const float* a_dst1 = (const float*)d_in[17];
  const float* b1     = (const float*)d_in[18];
  const float* W2     = (const float*)d_in[19];
  const float* a_src2 = (const float*)d_in[20];
  const float* a_dst2 = (const float*)d_in[21];
  const float* b2     = (const float*)d_in[22];
  const float* Wo1    = (const float*)d_in[23];
  const float* bo1    = (const float*)d_in[24];
  const float* Wo2    = (const float*)d_in[25];
  const float* bo2    = (const float*)d_in[26];
  float* out = (float*)d_out;

  // ---- workspace carve-up (floats) ----
  float* ws = (float*)d_ws;
  size_t off = 0;
  float* X    = ws + off; off += (size_t)NN * 128;
  float* XP1  = ws + off; off += (size_t)NN * 512;
  float* ACC1 = ws + off; off += (size_t)NN * 512;
  float* ALS1 = ws + off; off += (size_t)NN * 4;
  float* ALD1 = ws + off; off += (size_t)NN * 4;
  int*   MAX1 = (int*)(ws + off); off += (size_t)NN * 4;
  float* SUM1 = ws + off; off += (size_t)NN * 4;
  // conv2 overlays the (then dead) XP1 region:
  float* XP2  = XP1;
  float* ACC2 = XP1 + (size_t)NN * 128;
  float* ALS2 = XP1 + (size_t)NN * 256;
  float* ALD2 = ALS2 + NN;
  int*   MAX2 = (int*)(ALD2 + NN);
  float* SUM2 = ALD2 + 2 * (size_t)NN;
  float* POOL = ALD2 + 3 * (size_t)NN;
  float* H1b  = POOL + (size_t)GG * 128;
  // f16 weight copies (swizzled fragment order):
  _Float16* h16 = (_Float16*)(ws + off);
  size_t ho = 0;
  _Float16* WtS  = h16 + ho; ho += 768 * 128;
  _Float16* WvS  = h16 + ho; ho += 768 * 128;
  _Float16* WnS  = h16 + ho; ho += 128 * 128;
  _Float16* WsS  = h16 + ho; ho += 256 * 128;
  _Float16* W1S  = h16 + ho; ho += 128 * 512;
  _Float16* W2S  = h16 + ho; ho += 512 * 128;
  _Float16* Wo1S = h16 + ho; ho += 128 * 128;
  _Float16* embH = h16 + ho; ho += 51 * 128;

  auto cvtb = [&](const float* s, _Float16* d, int Ncols, int K) {
    k_cvt_b<<<(K * Ncols + 255) / 256, 256, 0, stream>>>(s, d, Ncols, K);
  };
  cvtb(Wt, WtS, 128, 768);     cvtb(Wv, WvS, 128, 768);
  cvtb(Wnode, WnS, 128, 128);  cvtb(Wstart, WsS, 128, 256);
  cvtb(W1, W1S, 512, 128);     cvtb(W2, W2S, 128, 512);
  cvtb(Wo1, Wo1S, 128, 128);
  k_cvt<<<(51 * 128 + 255) / 256, 256, 0, stream>>>(emb, embH, 51 * 128);

  // node embeddings @ Wnode -> X
  k_embed<<<dim3(NN / 16, 8), 32, 0, stream>>>(embH, m_idx, WnS, bnode, X);
  // start token: te/ve encoders + concat GEMM, overwrite node g*64
  k_start<<<GG / 16, 32, 0, stream>>>(t, v, WtS, bt, WvS, bv, WsS, bstart, X);
  // layernorm in place
  k_ln<<<NN / 8, 256, 0, stream>>>(X, ln_g, ln_b);

  // ---- GAT conv1 (4 heads, c=128) ----
  k_gemm<<<dim3(NN / 32, 16), 32, 0, stream>>>(X, W1S, nullptr, XP1, 512, 128, 0);
  k_dots<HEADS1><<<NN / 8, 256, 0, stream>>>(XP1, a_src1, a_dst1, ALS1, ALD1);
  k_init_max<HEADS1><<<NN * HEADS1 / 256, 256, 0, stream>>>(ALS1, ALD1, MAX1);
  k_edge_max<HEADS1><<<EE / 256, 256, 0, stream>>>(e_src, e_dst, ALS1, ALD1, MAX1);
  k_init_sum<HEADS1><<<NN * HEADS1 / 256, 256, 0, stream>>>(ALS1, ALD1, MAX1, SUM1);
  k_edge_sum<HEADS1><<<EE / 256, 256, 0, stream>>>(e_src, e_dst, ALS1, ALD1, MAX1, SUM1);
  k_init_acc<HEADS1><<<NN / 8, 256, 0, stream>>>(XP1, ALS1, ALD1, MAX1, SUM1, ACC1);
  k_edge_scatter<HEADS1><<<EE / 8, 256, 0, stream>>>(e_src, e_dst, XP1, ALS1, ALD1,
                                                     MAX1, SUM1, ACC1);
  k_bias_elu<<<(NN * 512) / 256, 256, 0, stream>>>(ACC1, b1, 512, NN * 512);

  // ---- GAT conv2 (1 head, c=128) ----
  k_gemm<<<dim3(NN / 32, 4), 32, 0, stream>>>(ACC1, W2S, nullptr, XP2, 128, 512, 0);
  k_dots<1><<<NN / 8, 256, 0, stream>>>(XP2, a_src2, a_dst2, ALS2, ALD2);
  k_init_max<1><<<NN / 256, 256, 0, stream>>>(ALS2, ALD2, MAX2);
  k_edge_max<1><<<EE / 256, 256, 0, stream>>>(e_src, e_dst, ALS2, ALD2, MAX2);
  k_init_sum<1><<<NN / 256, 256, 0, stream>>>(ALS2, ALD2, MAX2, SUM2);
  k_edge_sum<1><<<EE / 256, 256, 0, stream>>>(e_src, e_dst, ALS2, ALD2, MAX2, SUM2);
  k_init_acc<1><<<NN / 8, 256, 0, stream>>>(XP2, ALS2, ALD2, MAX2, SUM2, ACC2);
  k_edge_scatter<1><<<EE / 8, 256, 0, stream>>>(e_src, e_dst, XP2, ALS2, ALD2,
                                                MAX2, SUM2, ACC2);
  k_bias_elu<<<(NN * 128) / 256, 256, 0, stream>>>(ACC2, b2, 128, NN * 128);

  // ---- pool + head ----
  k_pool<<<GG / 8, 256, 0, stream>>>(ACC2, m_idx, POOL);
  k_gemm<<<dim3(GG / 32, 4), 32, 0, stream>>>(POOL, Wo1S, bo1, H1b, 128, 128, 2);
  k_logits<<<GG / 8, 256, 0, stream>>>(H1b, Wo2, bo2, out);
}
